// GeoUmpER_32693291057154
// MI455X (gfx1250) — matrix operand order repaired and checked by reading, not compiled, not run on hardware
//
#include <hip/hip_runtime.h>

// ---------------------------------------------------------------------------
// CDNA5 (gfx1250, wave32) implementation.
// Strategy: bf16 WMMA (v_wmma_f32_16x16x32_bf16) for all large GEMMs,
// double-buffered TDM (tensor_load_to_lds, TENSORcnt pipelining) for GEMM
// tile staging, f32 softmax/reductions.
// ---------------------------------------------------------------------------

typedef __attribute__((ext_vector_type(16))) __bf16 v16bf;
typedef __attribute__((ext_vector_type(8)))  float  v8f;
typedef unsigned int u32x4 __attribute__((ext_vector_type(4)));
typedef int          i32x8 __attribute__((ext_vector_type(8)));
typedef int          i32x4 __attribute__((ext_vector_type(4)));

#define DEVI __device__ __forceinline__

DEVI v8f wmma_bf16(v16bf a, v16bf b, v8f c) {
  // 8 args: (neg_a, A, neg_b, B, c_mod, C, reuse_a, reuse_b)
  return __builtin_amdgcn_wmma_f32_16x16x32_bf16(false, a, false, b, (short)0, c,
                                                 false, false);
}

// ---------------------------------------------------------------------------
// TDM: DMA one 2D tile (rows x 32 bf16 elements, row stride = stride_el) from
// global memory into LDS at byte offset lds_off. Descriptor per ISA ch.8:
//   group0: count=1 | lds_addr | global_addr[56:0] | type=2
//   group1: data_size=1(2B), tensor_dim0=stride_el, tensor_dim1=big,
//           tile_dim0=32, tile_dim1=rows, tensor_dim0_stride=stride_el
//   groups 2/3: zero (2D tensor)
// Issued once per wave; completion tracked with TENSORcnt (in-order per wave).
// ---------------------------------------------------------------------------
DEVI void tdm_load_tile_2d(unsigned lds_off, const void* gptr, unsigned rows,
                           unsigned stride_el) {
  const unsigned long long ga = (unsigned long long)gptr;
  u32x4 g0;
  g0[0] = 1u;                                        // count = 1, user mode
  g0[1] = lds_off;                                   // LDS byte address
  g0[2] = (unsigned)ga;                              // global_addr[31:0]
  g0[3] = (unsigned)((ga >> 32) & 0x01FFFFFFu) | 0x80000000u;  // addr hi | type=2
  const unsigned td0 = stride_el;                    // tensor_dim0 (elements)
  const unsigned td1 = 1u << 20;                     // tensor_dim1 (large)
  i32x8 g1;
  g1[0] = (int)(1u << 16);                           // data_size = 2 bytes
  g1[1] = (int)((td0 & 0xFFFFu) << 16);              // dim0 low16 << 16
  g1[2] = (int)((td0 >> 16) | ((td1 & 0xFFFFu) << 16));
  g1[3] = (int)((td1 >> 16) | (32u << 16));          // tile_dim0 = 32
  g1[4] = (int)(rows & 0xFFFFu);                     // tile_dim1 = rows
  g1[5] = (int)td0;                                  // tensor_dim0_stride lo
  g1[6] = 0;                                         // stride hi | dim1_stride lo
  g1[7] = 0;
  i32x4 gz = {0, 0, 0, 0};
#if defined(__clang_major__) && (__clang_major__ >= 23)
  i32x8 z8 = {0, 0, 0, 0, 0, 0, 0, 0};
  __builtin_amdgcn_tensor_load_to_lds(g0, g1, gz, gz, z8, 0);
#else
  __builtin_amdgcn_tensor_load_to_lds(g0, g1, gz, gz, 0);
#endif
}

// Low 32 bits of a flat LDS pointer == LDS byte offset (ISA 10.2 aperture map).
DEVI unsigned lds_offset_of(const void* p) {
  return (unsigned)(unsigned long long)p;
}

// A-fragment: 16x32 tile of a row-major bf16 matrix (ld in elements).
// Layout: lanes 0-15 row M=lane, K = {0..15}; lanes 16-31 same rows,
// K = {16..31}; VGPR p holds K pair (2p) per half.
DEVI v16bf frag_a(const __bf16* s, int ld) {
  const int lane = threadIdx.x & 31;
  const int row  = lane & 15;
  const int kb   = (lane & 16) ? 16 : 0;
  v16bf f;
#pragma unroll
  for (int p = 0; p < 8; ++p) {
    const int k = kb + 2 * p;
    f[2 * p]     = s[row * ld + k];
    f[2 * p + 1] = s[row * ld + k + 1];
  }
  return f;
}

// B-fragment: 32x16 (KxN) tile read from an [N][K] row-major matrix, so each
// lane reads 16 *contiguous* bf16: lane = column n (&15), K half by lane>=16.
DEVI v16bf frag_b(const __bf16* s, int ld) {
  const int lane = threadIdx.x & 31;
  const __bf16* p = s + (size_t)(lane & 15) * ld + ((lane & 16) ? 16 : 0);
  v16bf f;
#pragma unroll
  for (int e = 0; e < 16; ++e) f[e] = p[e];
  return f;
}

// ---------------------------------------------------------------------------
// Weight transpose + bf16 convert: src [K][N] f32 -> dst [N][K] bf16
// ---------------------------------------------------------------------------
__global__ __launch_bounds__(256) void transpose_w_bf16(
    const float* __restrict__ src, __bf16* __restrict__ dst, int K, int N) {
  const int idx = blockIdx.x * 256 + threadIdx.x;
  if (idx >= K * N) return;
  const int n = idx / K, k = idx % K;
  dst[idx] = (__bf16)src[(size_t)k * N + n];
}

// ---------------------------------------------------------------------------
// Pack e-features (8 rows, zero-padded to 16) as bf16: ef[set][b][16][768]
// ---------------------------------------------------------------------------
__global__ __launch_bounds__(256) void pack_efea(const float* __restrict__ xs,
                                                 __bf16* __restrict__ ef) {
  const int set = blockIdx.x;  // 0 -> rows 1..8, 1 -> rows 20..27
  const int b   = blockIdx.y;
  const int e0  = set ? 20 : 1;
  for (int i = threadIdx.x; i < 16 * 768; i += 256) {
    const int l = i / 768, h = i % 768;
    const float v = (l < 8) ? xs[((size_t)b * 512 + e0 + l) * 768 + h] : 0.0f;
    ef[((size_t)(set * 32 + b) * 16 + l) * 768 + h] = (__bf16)v;
  }
}

// ---------------------------------------------------------------------------
// Generic 64x128 tiled bf16 WMMA GEMM: C = A(MxK) * Wt(NxK)^T [+ bias]
//   A row-major bf16 (ld = K), Wt row-major [N][K] bf16.
//   Tile staging via double-buffered TDM: wave 0 issues the next (A,W) tile
//   pair, then s_wait_tensorcnt 2 -- TDM completes in order per wave, so <=2
//   outstanding means the current pair has landed; DMA overlaps WMMA.
//   8 waves: wave w -> mtile = w>>1 (0..3), ntiles (w&1)*4 + {0..3}.
//   TRANSC stores C[(m>>9)*N + n][m&511] (used to produce V^T per batch of 512).
// ---------------------------------------------------------------------------
template <bool BIAS, bool OUTBF, bool TRANSC>
__global__ __launch_bounds__(256) void gemm_bf16_wmma(
    const __bf16* __restrict__ A, const __bf16* __restrict__ Wt,
    const float* __restrict__ bias, void* __restrict__ Cv, int N, int K,
    long long aStride, long long wStride, long long cStride) {
  __shared__ __attribute__((aligned(64))) __bf16 As[2][64 * 32];
  __shared__ __attribute__((aligned(64))) __bf16 Ws[2][128 * 32];
  const int z = blockIdx.z;
  A  += (long long)z * aStride;
  Wt += (long long)z * wStride;
  const int m0   = blockIdx.x * 64;
  const int n0   = blockIdx.y * 128;
  const int tid  = threadIdx.x;
  const int wave = tid >> 5;
  const int lane = tid & 31;
  const int mt   = wave >> 1;
  const int ntb  = (wave & 1) * 4;
  const unsigned asOff0 = lds_offset_of(As[0]);
  const unsigned asOff1 = lds_offset_of(As[1]);
  const unsigned wsOff0 = lds_offset_of(Ws[0]);
  const unsigned wsOff1 = lds_offset_of(Ws[1]);
  v8f acc[4] = {};

  if (wave == 0) {  // prologue: DMA first tile pair into buffer 0
    tdm_load_tile_2d(asOff0, &A[(size_t)m0 * K], 64, (unsigned)K);
    tdm_load_tile_2d(wsOff0, &Wt[(size_t)n0 * K], 128, (unsigned)K);
  }
  int sel = 0;
  for (int kc = 0; kc < K; kc += 32) {
    if (wave == 0) {
      if (kc + 32 < K) {  // pipeline: issue next pair into the other buffer
        tdm_load_tile_2d(sel ? asOff0 : asOff1, &A[(size_t)m0 * K + kc + 32],
                         64, (unsigned)K);
        tdm_load_tile_2d(sel ? wsOff0 : wsOff1, &Wt[(size_t)n0 * K + kc + 32],
                         128, (unsigned)K);
        __builtin_amdgcn_s_wait_tensorcnt(2);  // current pair complete
      } else {
        __builtin_amdgcn_s_wait_tensorcnt(0);  // last pair complete
      }
    }
    __syncthreads();  // publish current buffer to all waves
    const v16bf af = frag_a(&As[sel][mt * 16 * 32], 32);
#pragma unroll
    for (int j = 0; j < 4; ++j) {
      const v16bf bf = frag_b(&Ws[sel][(ntb + j) * 16 * 32], 32);
      acc[j] = wmma_bf16(af, bf, acc[j]);
    }
    __syncthreads();  // all reads done before this buffer is DMA-overwritten
    sel ^= 1;
  }

  const int nl   = lane & 15;
  const int mrow = m0 + mt * 16 + ((lane & 16) ? 8 : 0);
#pragma unroll
  for (int j = 0; j < 4; ++j) {
    const int nj  = n0 + (ntb + j) * 16 + nl;
    const float bj = BIAS ? bias[nj] : 0.0f;
    if (OUTBF) {
      __bf16* C = (__bf16*)Cv + (long long)z * cStride;
#pragma unroll
      for (int r = 0; r < 8; ++r) {
        const int m   = mrow + r;
        const float v = acc[j][r] + bj;
        if (TRANSC) {
          C[((size_t)(m >> 9) * N + nj) * 512 + (m & 511)] = (__bf16)v;
        } else {
          C[(size_t)m * N + nj] = (__bf16)v;
        }
      }
    } else {
      float* C = (float*)Cv + (long long)z * cStride;
#pragma unroll
      for (int r = 0; r < 8; ++r) {
        const int m = mrow + r;
        C[(size_t)m * N + nj] = acc[j][r] + bj;
      }
    }
  }
}

// ---------------------------------------------------------------------------
// Fused neighbor-message kernel, one workgroup per (b,n):
//   Kn[32,256] = E[32,768] @ Wk^T + bk            (WMMA)
//   L[8,32]    = q[16pad,256] @ Kn^T * 1/16       (WMMA)
//   P          = softmax_t(L)                     (f32)
//   msg[8,768] = P @ E                            (WMMA)
// LDS: Ebf 48KB + Knb 16KB = 64KB; softmax scratch aliases dead Knb.
// ---------------------------------------------------------------------------
__global__ __launch_bounds__(256) void neighbor_msgs(
    const float* __restrict__ emb, const __bf16* __restrict__ wk_t,
    const float* __restrict__ bk, const __bf16* __restrict__ qpad,
    float* __restrict__ msg) {
  __shared__ __attribute__((aligned(64))) __bf16 Ebf[32 * 768];
  __shared__ __attribute__((aligned(64))) __bf16 Knb[32 * 256];
  const int bn   = blockIdx.x;
  const int b    = bn >> 5;
  const int tid  = threadIdx.x;
  const int wave = tid >> 5;
  const int lane = tid & 31;

  // Stage E (f32 -> bf16) into LDS
  const float* Eg = emb + (size_t)bn * 32 * 768;
  for (int i = tid; i < 32 * 768 / 4; i += 256) {
    const float4 v = ((const float4*)Eg)[i];
    Ebf[4 * i + 0] = (__bf16)v.x;
    Ebf[4 * i + 1] = (__bf16)v.y;
    Ebf[4 * i + 2] = (__bf16)v.z;
    Ebf[4 * i + 3] = (__bf16)v.w;
  }
  __syncthreads();

  // Kn = E @ Wk^T + bk : M=32 (2 mtiles), N=256 (16 ntiles), K=768
  {
    const int mtile = wave & 1;
    const int ntq   = wave >> 1;  // 4 ntiles per wave
    v8f acc[4] = {};
    for (int kc = 0; kc < 768; kc += 32) {
      const v16bf af = frag_a(&Ebf[mtile * 16 * 768 + kc], 768);
#pragma unroll
      for (int j = 0; j < 4; ++j) {
        const int nt   = ntq * 4 + j;
        const v16bf bf = frag_b(&wk_t[(size_t)nt * 16 * 768 + kc], 768);
        acc[j] = wmma_bf16(af, bf, acc[j]);
      }
      if (kc + 32 < 768)  // gfx1250: lowers to global_prefetch_b8
        __builtin_prefetch(&wk_t[(size_t)(ntq * 4) * 16 * 768 + kc + 32]);
    }
#pragma unroll
    for (int j = 0; j < 4; ++j) {
      const int n    = (ntq * 4 + j) * 16 + (lane & 15);
      const float bb = bk[n];
      const int mb   = mtile * 16 + ((lane & 16) ? 8 : 0);
#pragma unroll
      for (int r = 0; r < 8; ++r) Knb[(mb + r) * 256 + n] = (__bf16)(acc[j][r] + bb);
    }
  }
  __syncthreads();

  // logits: waves 0,1 each produce one 16x16 tile (t-columns 0..15 / 16..31)
  v8f lacc = {};
  if (wave < 2) {
    const __bf16* Q = qpad + (size_t)b * 16 * 256;  // caller pre-offsets set
    for (int kc = 0; kc < 256; kc += 32) {
      const v16bf af = frag_a(&Q[kc], 256);
      const v16bf bf = frag_b(&Knb[wave * 16 * 256 + kc], 256);
      lacc = wmma_bf16(af, bf, lacc);
    }
  }
  __syncthreads();
  float*  Lf = (float*)Knb;               // Knb dead: alias 16x32 f32 logits
  __bf16* Pb = (__bf16*)(Lf + 16 * 32);   // 16x32 bf16 probs
  if (wave < 2) {
    const int t  = wave * 16 + (lane & 15);
    const int mb = (lane & 16) ? 8 : 0;
#pragma unroll
    for (int r = 0; r < 8; ++r) Lf[(mb + r) * 32 + t] = lacc[r] * 0.0625f;
  }
  __syncthreads();
  if (tid < 16) {
    if (tid < 8) {
      float mx = -1e30f;
      for (int t = 0; t < 32; ++t) mx = fmaxf(mx, Lf[tid * 32 + t]);
      float s = 0.0f;
      for (int t = 0; t < 32; ++t) s += __expf(Lf[tid * 32 + t] - mx);
      const float inv = 1.0f / s;
      for (int t = 0; t < 32; ++t)
        Pb[tid * 32 + t] = (__bf16)(__expf(Lf[tid * 32 + t] - mx) * inv);
    } else {
      for (int t = 0; t < 32; ++t) Pb[tid * 32 + t] = (__bf16)0.0f;
    }
  }
  __syncthreads();

  // msg = P(16x32) @ E(32x768): 48 column tiles, 6 per wave, K=32 single step
  {
    const v16bf af = frag_a(Pb, 32);
    float* out   = msg + (size_t)bn * 8 * 768;
    const int h  = lane & 15;
    const int kb = (lane & 16) ? 16 : 0;
#pragma unroll
    for (int j = 0; j < 6; ++j) {
      const int nt = wave * 6 + j;
      const int hc = nt * 16 + h;
      v16bf bf;
#pragma unroll
      for (int e = 0; e < 16; ++e) bf[e] = Ebf[(kb + e) * 768 + hc];
      v8f acc = {};
      acc = wmma_bf16(af, bf, acc);
      if (!(lane & 16)) {  // only rows 0..7 are real message rows
#pragma unroll
        for (int r = 0; r < 8; ++r) out[(size_t)r * 768 + hc] = acc[r];
      }
    }
  }
}

// ---------------------------------------------------------------------------
// cal_attn: per (b, set). score_n = leaky(wf.W1 + wp_n.W2 + battn) + d*wb + bb,
// softmax over n. Tiny -> plain VALU.
// ---------------------------------------------------------------------------
__global__ __launch_bounds__(256) void cal_attn_kernel(
    const float* __restrict__ xs, const float* __restrict__ msg,
    const float* __restrict__ dist, const float* __restrict__ Wa,
    const float* __restrict__ ba, const float* __restrict__ Wattn,
    const float* __restrict__ battn, const float* __restrict__ wb,
    const float* __restrict__ bb, float* __restrict__ att, int e0) {
  __shared__ float pooled[768];
  __shared__ float red[256];
  __shared__ float score[32];
  __shared__ float wfdot;
  const int b   = blockIdx.x;
  const int tid = threadIdx.x;

  for (int h = tid; h < 768; h += 256) {
    float s = 0.0f;
    for (int l = 0; l < 8; ++l) s += xs[((size_t)b * 512 + e0 + l) * 768 + h];
    pooled[h] = s * 0.125f;
  }
  __syncthreads();
  float part = 0.0f;
  if (tid < 128) {
    float wf = ba[tid];
    for (int h = 0; h < 768; ++h) wf += pooled[h] * Wa[(size_t)h * 128 + tid];
    part = wf * Wattn[tid];
  }
  red[tid] = part;
  __syncthreads();
  for (int s = 128; s > 0; s >>= 1) {
    if (tid < s) red[tid] += red[tid + s];
    __syncthreads();
  }
  if (tid == 0) wfdot = red[0];
  __syncthreads();

  for (int n = 0; n < 32; ++n) {
    for (int h = tid; h < 768; h += 256) {
      float s = 0.0f;
      for (int l = 0; l < 8; ++l)
        s += msg[(((size_t)(b * 32 + n)) * 8 + l) * 768 + h];
      pooled[h] = s * 0.125f;
    }
    __syncthreads();
    float p2 = 0.0f;
    if (tid < 128) {
      float wp = ba[tid];
      for (int h = 0; h < 768; ++h) wp += pooled[h] * Wa[(size_t)h * 128 + tid];
      p2 = wp * Wattn[128 + tid];
    }
    red[tid] = p2;
    __syncthreads();
    for (int s = 128; s > 0; s >>= 1) {
      if (tid < s) red[tid] += red[tid + s];
      __syncthreads();
    }
    if (tid == 0) {
      const float z  = wfdot + red[0] + battn[0];
      const float lk = (z >= 0.0f) ? z : 0.01f * z;
      score[n] = lk + dist[b * 32 + n] * wb[0] + bb[0];
    }
    __syncthreads();
  }
  if (tid == 0) {
    float mx = -1e30f;
    for (int n = 0; n < 32; ++n) mx = fmaxf(mx, score[n]);
    float s = 0.0f;
    for (int n = 0; n < 32; ++n) s += __expf(score[n] - mx);
    const float inv = 1.0f / s;
    for (int n = 0; n < 32; ++n) att[b * 32 + n] = __expf(score[n] - mx) * inv;
  }
}

// ---------------------------------------------------------------------------
// Build updated x in bf16: x += sum_n att[n]*msg[n] on the two edge windows.
// ---------------------------------------------------------------------------
__global__ __launch_bounds__(256) void build_x(
    const float* __restrict__ xs, const float* __restrict__ msg1,
    const float* __restrict__ msg2, const float* __restrict__ att1,
    const float* __restrict__ att2, __bf16* __restrict__ xbf) {
  const int b = blockIdx.x, s = blockIdx.y;
  for (int h = threadIdx.x; h < 768; h += 256) {
    float v = xs[((size_t)b * 512 + s) * 768 + h];
    if (s >= 1 && s < 9) {
      const int l = s - 1;
      float a = 0.0f;
      for (int n = 0; n < 32; ++n)
        a += att1[b * 32 + n] * msg1[(((size_t)(b * 32 + n)) * 8 + l) * 768 + h];
      v += a;
    } else if (s >= 20 && s < 28) {
      const int l = s - 20;
      float a = 0.0f;
      for (int n = 0; n < 32; ++n)
        a += att2[b * 32 + n] * msg2[(((size_t)(b * 32 + n)) * 8 + l) * 768 + h];
      v += a;
    }
    xbf[((size_t)b * 512 + s) * 768 + h] = (__bf16)v;
  }
}

// ---------------------------------------------------------------------------
// Row softmax over 512 logits (scaled 1/16), one wave32 per row, bf16 out.
// ---------------------------------------------------------------------------
__global__ __launch_bounds__(256) void softmax_rows(
    const float* __restrict__ logits, __bf16* __restrict__ P) {
  const int row  = blockIdx.x * 8 + (threadIdx.x >> 5);
  const int lane = threadIdx.x & 31;
  const float* L = logits + (size_t)row * 512;
  float v[16];
  float mx = -1e30f;
#pragma unroll
  for (int i = 0; i < 16; ++i) {
    v[i] = L[lane + i * 32] * 0.0625f;
    mx   = fmaxf(mx, v[i]);
  }
#pragma unroll
  for (int o = 16; o; o >>= 1) mx = fmaxf(mx, __shfl_xor(mx, o, 32));
  float s = 0.0f;
#pragma unroll
  for (int i = 0; i < 16; ++i) {
    v[i] = __expf(v[i] - mx);
    s += v[i];
  }
#pragma unroll
  for (int o = 16; o; o >>= 1) s += __shfl_xor(s, o, 32);
  const float inv = 1.0f / s;
  __bf16* Po = P + (size_t)row * 512;
#pragma unroll
  for (int i = 0; i < 16; ++i) Po[lane + i * 32] = (__bf16)(v[i] * inv);
}

// ---------------------------------------------------------------------------
// Host launcher
// ---------------------------------------------------------------------------
extern "C" void kernel_launch(void* const* d_in, const int* in_sizes, int n_in,
                              void* d_out, int out_size, void* d_ws,
                              size_t ws_size, hipStream_t stream) {
  (void)in_sizes; (void)n_in; (void)out_size; (void)ws_size;
  const float* xs    = (const float*)d_in[0];
  const float* n1emb = (const float*)d_in[1];
  const float* n2emb = (const float*)d_in[2];
  const float* dist1 = (const float*)d_in[3];
  const float* dist2 = (const float*)d_in[4];
  const float* Wq    = (const float*)d_in[5];
  const float* bq    = (const float*)d_in[6];
  const float* Wk    = (const float*)d_in[7];
  const float* bk    = (const float*)d_in[8];
  const float* Wv    = (const float*)d_in[9];
  const float* bv    = (const float*)d_in[10];
  const float* Wa    = (const float*)d_in[11];
  const float* ba    = (const float*)d_in[12];
  const float* Wattn = (const float*)d_in[13];
  const float* battn = (const float*)d_in[14];
  const float* wb    = (const float*)d_in[15];
  const float* bb    = (const float*)d_in[16];
  float* out = (float*)d_out;

  // Workspace layout (bytes, every size multiple of 256)
  char* w = (char*)d_ws;
  size_t o = 0;
  auto take = [&](size_t bytes) { char* p = w + o; o += bytes; return p; };
  __bf16* wq_t  = (__bf16*)take(256 * 768 * 2);            // [256][768]
  __bf16* wk_t  = (__bf16*)take(256 * 768 * 2);            // [256][768]
  __bf16* wv_t  = (__bf16*)take(768 * 768 * 2);            // [768][768]
  __bf16* efea  = (__bf16*)take(2ull * 32 * 16 * 768 * 2); // [set][b][16][768]
  __bf16* qpad  = (__bf16*)take(2ull * 32 * 16 * 256 * 2); // [set][b][16][256]
  float*  msg1  = (float*)take(32ull * 32 * 8 * 768 * 4);
  float*  msg2  = (float*)take(32ull * 32 * 8 * 768 * 4);
  float*  att1  = (float*)take(32 * 32 * 4);
  float*  att2  = (float*)take(32 * 32 * 4);
  __bf16* xbf   = (__bf16*)take(32ull * 512 * 768 * 2);
  __bf16* qb    = (__bf16*)take(32ull * 512 * 256 * 2);
  __bf16* kb    = (__bf16*)take(32ull * 512 * 256 * 2);
  __bf16* vt    = (__bf16*)take(32ull * 768 * 512 * 2);    // [b][h][t]
  __bf16* pbf   = (__bf16*)take(32ull * 512 * 512 * 2);
  // logits (f32, 33.5MB) alias the msg region (50.3MB) which is dead by then.
  float*  logits = (float*)msg1;

  // 1) weights -> bf16 [N][K]
  transpose_w_bf16<<<(768 * 256 + 255) / 256, 256, 0, stream>>>(Wq, wq_t, 768, 256);
  transpose_w_bf16<<<(768 * 256 + 255) / 256, 256, 0, stream>>>(Wk, wk_t, 768, 256);
  transpose_w_bf16<<<(768 * 768 + 255) / 256, 256, 0, stream>>>(Wv, wv_t, 768, 768);

  // 2) e-features (bf16, padded) and q = e_fea @ Wq + bq
  pack_efea<<<dim3(2, 32), 256, 0, stream>>>(xs, efea);
  gemm_bf16_wmma<true, true, false><<<dim3(16, 2, 1), 256, 0, stream>>>(
      efea, wq_t, bq, qpad, 256, 768, 0, 0, 0);

  // 3) fused neighbor messages (WMMA Kn, WMMA logits, softmax, WMMA msg)
  neighbor_msgs<<<1024, 256, 0, stream>>>(n1emb, wk_t, bk, qpad, msg1);
  neighbor_msgs<<<1024, 256, 0, stream>>>(n2emb, wk_t, bk,
                                          qpad + 32ull * 16 * 256, msg2);

  // 4) attention weights over neighborhoods, then updated x (bf16)
  cal_attn_kernel<<<32, 256, 0, stream>>>(xs, msg1, dist1, Wa, ba, Wattn, battn,
                                          wb, bb, att1, 1);
  cal_attn_kernel<<<32, 256, 0, stream>>>(xs, msg2, dist2, Wa, ba, Wattn, battn,
                                          wb, bb, att2, 20);
  build_x<<<dim3(32, 512), 256, 0, stream>>>(xs, msg1, msg2, att1, att2, xbf);

  // 5) QKV projections (M = 32*512 = 16384)
  gemm_bf16_wmma<true, true, false><<<dim3(256, 2, 1), 256, 0, stream>>>(
      xbf, wq_t, bq, qb, 256, 768, 0, 0, 0);
  gemm_bf16_wmma<true, true, false><<<dim3(256, 2, 1), 256, 0, stream>>>(
      xbf, wk_t, bk, kb, 256, 768, 0, 0, 0);
  gemm_bf16_wmma<true, true, true><<<dim3(256, 6, 1), 256, 0, stream>>>(
      xbf, wv_t, bv, vt, 768, 768, 0, 0, 0);  // stores V^T per 512-row batch

  // 6) logits = Q @ K^T (batched over b), softmax, out = P @ V
  gemm_bf16_wmma<false, false, false><<<dim3(8, 4, 32), 256, 0, stream>>>(
      qb, kb, nullptr, logits, 512, 256, 512ll * 256, 512ll * 256, 512ll * 512);
  softmax_rows<<<2048, 256, 0, stream>>>(logits, pbf);
  gemm_bf16_wmma<false, false, false><<<dim3(8, 6, 32), 256, 0, stream>>>(
      pbf, vt, nullptr, out, 768, 512, 512ll * 512, 768ll * 512, 512ll * 768);
}
